// ImageTriplaneGenerator_42666205118759
// MI455X (gfx1250) — matrix operand order repaired and checked by reading, not compiled, not run on hardware
//
#include <hip/hip_runtime.h>

// ---------------- problem constants (from reference) ----------------
constexpr int NV_ = 4, C_ = 196, H_ = 128, W_ = 128;
constexpr int G_ = 128, S_ = 8, NL_ = 3;
constexpr int L_ = G_ * G_;           // 16384
constexpr int F_ = 4 * C_;            // 784
constexpr int M_ = L_;                // GEMM rows
// padded dims: multiples of 32 (WMMA K-step) and 64 (N block)
constexpr int CP = 256;
constexpr int FP = 832;

typedef __attribute__((ext_vector_type(16))) _Float16 v16h;
typedef __attribute__((ext_vector_type(8)))  float    v8f;

union Frag { v16h h; unsigned u[8]; };

// ---------------- utility kernels ----------------
__global__ void zero_f32(float* __restrict__ p, long n) {
  long id = (long)blockIdx.x * blockDim.x + threadIdx.x;
  if (id < n) p[id] = 0.0f;
}

__global__ void pad_f32_kernel(const float* __restrict__ src, float* __restrict__ dst,
                               int srcRows, int srcCols, int dstCols, long total) {
  long id = (long)blockIdx.x * blockDim.x + threadIdx.x;
  if (id >= total) return;
  long r = id / dstCols; int c = (int)(id % dstCols);
  dst[id] = (r < srcRows && c < srcCols) ? src[r * (long)srcCols + c] : 0.0f;
}

__global__ void pad_f16_kernel(const float* __restrict__ src, _Float16* __restrict__ dst,
                               int srcRows, int srcCols, int dstCols, long total) {
  long id = (long)blockIdx.x * blockDim.x + threadIdx.x;
  if (id >= total) return;
  long r = id / dstCols; int c = (int)(id % dstCols);
  dst[id] = (r < srcRows && c < srcCols) ? (_Float16)src[r * (long)srcCols + c] : (_Float16)0.0f;
}

__global__ void pad_bias_kernel(const float* __restrict__ src, float* __restrict__ dst,
                                int n, int np) {
  int c = blockIdx.x * blockDim.x + threadIdx.x;
  if (c < np) dst[c] = (c < n) ? src[c] : 0.0f;
}

__global__ void copy_out_kernel(const float* __restrict__ qf, float* __restrict__ out) {
  long id = (long)blockIdx.x * blockDim.x + threadIdx.x;
  if (id >= 3L * L_ * C_) return;
  long pl = id / C_; int c = (int)(id % C_);
  out[id] = qf[pl * CP + c];
}

// ---------------- phase 1: view-context scatter ----------------
__global__ void scatter_kernel(const float* __restrict__ feat,   // [C][H][W] for one view
                               const float* __restrict__ dep,    // [H][W]
                               const float* __restrict__ c2w,    // [4][4]
                               const float* __restrict__ Kin,    // [3][3]
                               float* __restrict__ vsum,         // [3][L][C]
                               float* __restrict__ vcnt) {       // [3][L]
  const int pix = blockIdx.x * blockDim.x + threadIdx.x;
  if (pix >= H_ * W_) return;
  const int i = pix / W_, j = pix % W_;
  float d = dep[pix];
  if (d == 0.0f) return;              // masked pixel contributes nothing
  d += 0.01f;
  // inverse of 3x3 intrinsic
  const float a = Kin[0], b = Kin[1], cc = Kin[2];
  const float e = Kin[3], f = Kin[4], g  = Kin[5];
  const float h = Kin[6], k = Kin[7], m  = Kin[8];
  const float det = a * (f * m - g * k) - b * (e * m - g * h) + cc * (e * k - f * h);
  const float inv = 1.0f / det;
  const float i00 = (f * m - g * k) * inv, i01 = (cc * k - b * m) * inv, i02 = (b * g - cc * f) * inv;
  const float i10 = (g * h - e * m) * inv, i11 = (a * m - cc * h) * inv, i12 = (cc * e - a * g) * inv;
  const float i20 = (e * k - f * h) * inv, i21 = (b * h - a * k) * inv, i22 = (a * f - b * e) * inv;
  const float u = (float)j, vv = (float)i;
  const float cx = (i00 * u + i01 * vv + i02) * d;
  const float cy = (i10 * u + i11 * vv + i12) * d;
  const float cz = (i20 * u + i21 * vv + i22) * d;
  const float wx = c2w[0] * cx + c2w[1] * cy + c2w[2]  * cz + c2w[3];
  const float wy = c2w[4] * cx + c2w[5] * cy + c2w[6]  * cz + c2w[7];
  const float wz = c2w[8] * cx + c2w[9] * cy + c2w[10] * cz + c2w[11];
  if (wx < -1.0f || wx > 1.0f || wy < -1.0f || wy > 1.0f || wz < -1.0f || wz > 1.0f) return;
  const float w3[3] = {wx, wy, wz};
  const int A0[3] = {0, 0, 1}, A1[3] = {1, 2, 2};
  for (int p = 0; p < 3; ++p) {
    const float c0 = w3[A0[p]], c1 = w3[A1[p]];
    const int px = (int)fminf(fmaxf(rintf((c0 * 0.5f + 0.5f) * (G_ - 1)), 0.0f), (float)(G_ - 1));
    const int py = (int)fminf(fmaxf(rintf((c1 * 0.5f + 0.5f) * (G_ - 1)), 0.0f), (float)(G_ - 1));
    const long cell = (long)p * L_ + (long)py * G_ + px;
    atomicAdd(&vcnt[cell], 1.0f);
    float* dst = &vsum[cell * C_];
    for (int c = 0; c < C_; ++c) atomicAdd(&dst[c], feat[(size_t)c * H_ * W_ + pix]);
  }
}

__global__ void ctx_accum_kernel(const float* __restrict__ vsum, const float* __restrict__ vcnt,
                                 float* __restrict__ ctx_acc) {
  long id = (long)blockIdx.x * blockDim.x + threadIdx.x;
  if (id >= 3L * L_ * C_) return;
  long cell = id / C_;
  ctx_acc[id] += vsum[id] / fmaxf(vcnt[cell], 1.0f) * (1.0f / (float)NV_);
}

// ---------------- WMMA GEMM: out[M][Np] = A[M][Kp] * B[Kp][Np] + bias ----------------
// 8 wave32 waves; block tile 256(M) x 64(N); each wave owns TWO 16-row strips so every
// B fragment feeds 2 WMMAs (8 v_wmma per K-step). K-step 32; f16 in, f32 accumulate.
// Double-buffered LDS; A tile staged with CDNA5 GLOBAL_LOAD_ASYNC_TO_LDS (ASYNCcnt),
// B tile staged through VGPRs because it is transposed on the way into LDS.
__global__ __launch_bounds__(256) void gemm_wmma(const _Float16* __restrict__ A,
                                                 const _Float16* __restrict__ Bw,
                                                 const float* __restrict__ bias,
                                                 float* __restrict__ outF,
                                                 _Float16* __restrict__ outH,
                                                 int Kp, int Np, int doGelu) {
  __shared__ _Float16 As[2][256 * 40];   // row-major, stride 40 halves
  __shared__ _Float16 Bs[2][64 * 34];    // transposed [colLocal][K], stride 34 halves
  __builtin_amdgcn_s_wait_tensorcnt(0);

  const int t    = threadIdx.x;
  const int wave = t >> 5;
  const int lane = t & 31;
  const int grp  = lane >> 4;
  const int ln16 = lane & 15;
  const long rowBase = (long)blockIdx.y * 256;
  const int  colBase = blockIdx.x * 64;

  v8f acc[2][4];
  for (int s = 0; s < 2; ++s)
    for (int n = 0; n < 4; ++n)
      for (int r = 0; r < 8; ++r) acc[s][n][r] = 0.0f;

  const int nK    = Kp >> 5;
  const int b_row = t >> 3;           // 0..31
  const int b_seg = (t & 7) * 8;      // local col 0..56

  const _Float16* aBase = A + (rowBase + t) * Kp;            // one 32-half row per thread
  const _Float16* bBase = Bw + (long)b_row * Np + colBase + b_seg;

  // async copy of this thread's 64B row of the A tile (kk-th K step) into buffer `buf`
  auto stageA = [&](int buf, int kk) {
    // low 32 bits of a generic LDS pointer are the LDS byte offset (aperture layout)
    unsigned l0 = (unsigned)(size_t)&As[buf][t * 40];
    unsigned long long ga = (unsigned long long)(size_t)(aBase + (kk << 5));
    #pragma unroll
    for (int q = 0; q < 4; ++q) {
      asm volatile("global_load_async_to_lds_b128 %0, %1, off"
                   :: "v"(l0 + 16u * q), "v"(ga + 16ull * q) : "memory");
    }
  };

  // ---- prologue: stage tile 0 into buffer 0 ----
  stageA(0, 0);
  {
    uint4 d = *(const uint4*)(bBase);
    const _Float16* hv = (const _Float16*)&d;
    #pragma unroll
    for (int q = 0; q < 8; ++q) Bs[0][(b_seg + q) * 34 + b_row] = hv[q];
  }
  asm volatile("s_wait_asynccnt 0x0" ::: "memory");
  __syncthreads();

  for (int kk = 0; kk < nK; ++kk) {
    const int buf = kk & 1, nbuf = buf ^ 1;
    const bool more = (kk + 1 < nK);
    uint4 bnext = {};
    if (more) {
      stageA(nbuf, kk + 1);                                   // async A for next step
      bnext = *(const uint4*)(bBase + (long)(kk + 1) * 32 * Np);
      if (kk + 2 < nK)                                        // near-scope prefetch
        __builtin_prefetch(bBase + (long)(kk + 2) * 32 * Np, 0, 3);
    }

    // A fragments: documented 16-bit A 16x32 lane layout -> two contiguous 16B runs
    Frag fa[2];
    #pragma unroll
    for (int s = 0; s < 2; ++s) {
      const _Float16* basea = &As[buf][(wave * 32 + s * 16 + ln16) * 40];
      #pragma unroll
      for (int vg = 0; vg < 8; ++vg) {
        const int K = (vg < 4) ? (8 * grp + 2 * vg) : (16 + 8 * grp + 2 * (vg - 4));
        fa[s].u[vg] = *(const unsigned*)(basea + K);
      }
    }
    #pragma unroll
    for (int sub = 0; sub < 4; ++sub) {
      Frag fb;
      const _Float16* baseb = &Bs[buf][(sub * 16 + ln16) * 34];
      #pragma unroll
      for (int vg = 0; vg < 8; ++vg) fb.u[vg] = *(const unsigned*)(baseb + ((vg < 4) ? (8 * grp + 2 * vg) : (16 + 8 * grp + 2 * (vg - 4))));
      acc[0][sub] = __builtin_amdgcn_wmma_f32_16x16x32_f16(
          false, fa[0].h, false, fb.h, (short)0, acc[0][sub], false, false);
      acc[1][sub] = __builtin_amdgcn_wmma_f32_16x16x32_f16(
          false, fa[1].h, false, fb.h, (short)0, acc[1][sub], false, false);
    }

    if (more) { // transpose-store next B tile, then make next buffers visible
      const _Float16* hv = (const _Float16*)&bnext;
      #pragma unroll
      for (int q = 0; q < 8; ++q) Bs[nbuf][(b_seg + q) * 34 + b_row] = hv[q];
      asm volatile("s_wait_asynccnt 0x0" ::: "memory");
    }
    __syncthreads();
  }

  // epilogue: C/D layout -> VGPR r holds row r (lanes 0-15) / r+8 (lanes 16-31)
  #pragma unroll
  for (int s = 0; s < 2; ++s) {
    const long orow0 = rowBase + wave * 32 + s * 16 + 8 * grp;
    #pragma unroll
    for (int sub = 0; sub < 4; ++sub) {
      const int col = colBase + sub * 16 + ln16;
      const float bv = bias[col];
      #pragma unroll
      for (int r = 0; r < 8; ++r) {
        const long row = orow0 + r;
        float v = acc[s][sub][r] + bv;
        if (doGelu) v = 0.5f * v * (1.0f + erff(v * 0.70710678118f));
        if (outH) outH[row * Np + col] = (_Float16)v;
        else      outF[row * Np + col] = v;
      }
    }
  }
}

// ---------------- LayerNorm kernels (one block = one row; wave32 reductions) ---------
__device__ __forceinline__ float block_sum(float v, float* red) {
  #pragma unroll
  for (int o = 16; o > 0; o >>= 1) v += __shfl_down(v, o, 32);
  if ((threadIdx.x & 31) == 0) red[threadIdx.x >> 5] = v;
  __syncthreads();
  if (threadIdx.x == 0) {
    float s = 0.0f;
    #pragma unroll
    for (int i = 0; i < 8; ++i) s += red[i];
    red[0] = s;
  }
  __syncthreads();
  float r = red[0];
  __syncthreads();
  return r;
}

// qn = LN(q); q2 = q + qn*k*(1/7)*v; q = LN(q2, n1); writes f32 + f16 copies
__global__ __launch_bounds__(256) void ln_mid_kernel(
    float* __restrict__ qf, _Float16* __restrict__ qh,
    const float* __restrict__ kb, const float* __restrict__ vb,
    const float* __restrict__ lnw, const float* __restrict__ lnb,
    const float* __restrict__ n1w, const float* __restrict__ n1b) {
  __shared__ float red[8];
  const int l = blockIdx.x, c = threadIdx.x;
  float* qrow = qf + (size_t)l * CP;
  const bool act = (c < C_);
  const float x = act ? qrow[c] : 0.0f;
  const float mean = block_sum(x, red) * (1.0f / C_);
  const float dx = act ? (x - mean) : 0.0f;
  const float var = block_sum(dx * dx, red) * (1.0f / C_);
  const float qn = act ? (dx * rsqrtf(var + 1e-5f) * lnw[c] + lnb[c]) : 0.0f;
  const float kv = act ? kb[(size_t)l * CP + c] : 0.0f;
  const float vv = act ? vb[(size_t)l * CP + c] : 0.0f;
  const float q2 = x + qn * kv * (1.0f / 7.0f) * vv;   // scale = 1/sqrt(C//NH)=1/7
  const float m2 = block_sum(act ? q2 : 0.0f, red) * (1.0f / C_);
  const float d2 = act ? (q2 - m2) : 0.0f;
  const float v2 = block_sum(d2 * d2, red) * (1.0f / C_);
  const float out = act ? (d2 * rsqrtf(v2 + 1e-5f) * n1w[c] + n1b[c]) : 0.0f;
  qrow[c] = out;
  qh[(size_t)l * CP + c] = (_Float16)out;
}

// q = LN(q + h, n2)
__global__ __launch_bounds__(256) void ln_post_kernel(
    float* __restrict__ qf, const float* __restrict__ hb,
    const float* __restrict__ n2w, const float* __restrict__ n2b) {
  __shared__ float red[8];
  const int l = blockIdx.x, c = threadIdx.x;
  float* qrow = qf + (size_t)l * CP;
  const bool act = (c < C_);
  const float q2 = act ? (qrow[c] + hb[(size_t)l * CP + c]) : 0.0f;
  const float m = block_sum(q2, red) * (1.0f / C_);
  const float d = act ? (q2 - m) : 0.0f;
  const float v = block_sum(d * d, red) * (1.0f / C_);
  const float out = act ? (d * rsqrtf(v + 1e-5f) * n2w[c] + n2b[c]) : 0.0f;
  qrow[c] = out;
}

// ---------------- cross-plane bilinear context ----------------
__global__ __launch_bounds__(256) void ctxcp_kernel(const float* __restrict__ qf,
                                                    _Float16* __restrict__ out) {
  const int l = blockIdx.x;
  const int p = blockIdx.y;
  const int c = threadIdx.x;
  __shared__ int   sx0[16], sx1[16], sy0[16], sy1[16], so[16];
  __shared__ float swx[16], swy[16];
  if (threadIdx.x < 16) {
    const int s16 = threadIdx.x;
    const int oidx = s16 >> 3;     // which of the two other planes
    const int s = s16 & 7;         // depth slice
    int o = 0, seen = 0;
    for (int tt = 0; tt < 3; ++tt) { if (tt == p) continue; if (seen == oidx) o = tt; ++seen; }
    const int i = l / G_, j = l % G_;
    const float gx = -1.0f + 2.0f * j / (G_ - 1);
    const float gy = -1.0f + 2.0f * i / (G_ - 1);
    const float z  = -1.0f + 2.0f * s / (S_ - 1);
    float vec[3];
    if (p == 0)      { vec[0] = gx; vec[1] = gy; vec[2] = z;  }
    else if (p == 1) { vec[0] = gx; vec[1] = z;  vec[2] = gy; }
    else             { vec[0] = z;  vec[1] = gx; vec[2] = gy; }
    const int A0[3] = {0, 0, 1}, A1[3] = {1, 2, 2};
    const float px = fminf(fmaxf((vec[A0[o]] * 0.5f + 0.5f) * (G_ - 1), 0.0f), (float)(G_ - 1));
    const float py = fminf(fmaxf((vec[A1[o]] * 0.5f + 0.5f) * (G_ - 1), 0.0f), (float)(G_ - 1));
    const float x0 = floorf(px), y0 = floorf(py);
    sx0[s16] = (int)x0; sx1[s16] = (int)fminf(x0 + 1.0f, (float)(G_ - 1));
    sy0[s16] = (int)y0; sy1[s16] = (int)fminf(y0 + 1.0f, (float)(G_ - 1));
    swx[s16] = px - x0; swy[s16] = py - y0; so[s16] = o;
  }
  __syncthreads();
  float accv = 0.0f;
  if (c < C_) {
    #pragma unroll 4
    for (int s16 = 0; s16 < 16; ++s16) {
      const float* g = qf + (size_t)so[s16] * L_ * CP;
      const float wx = swx[s16], wy = swy[s16];
      const float f00 = g[((long)sy0[s16] * G_ + sx0[s16]) * CP + c];
      const float f01 = g[((long)sy0[s16] * G_ + sx1[s16]) * CP + c];
      const float f10 = g[((long)sy1[s16] * G_ + sx0[s16]) * CP + c];
      const float f11 = g[((long)sy1[s16] * G_ + sx1[s16]) * CP + c];
      accv += (f00 * (1.0f - wx) + f01 * wx) * (1.0f - wy)
            + (f10 * (1.0f - wx) + f11 * wx) * wy;
    }
    accv *= (1.0f / 16.0f);
  }
  out[((size_t)p * L_ + l) * CP + c] = (_Float16)accv;
}

// ---------------- host orchestration ----------------
extern "C" void kernel_launch(void* const* d_in, const int* in_sizes, int n_in,
                              void* d_out, int out_size, void* d_ws, size_t ws_size,
                              hipStream_t stream) {
  (void)in_sizes; (void)n_in; (void)out_size; (void)ws_size;
  const float* feats = (const float*)d_in[0];
  const float* dep   = (const float*)d_in[1];
  const float* c2w   = (const float*)d_in[2];
  const float* intr  = (const float*)d_in[3];
  const float* qpl   = (const float*)d_in[4];
  const float* cvp[14]; const float* cpp[14];
  for (int k = 0; k < 14; ++k) { cvp[k] = (const float*)d_in[5 + k]; cpp[k] = (const float*)d_in[19 + k]; }
  enum { LNW = 0, LNB, WK, BK, WV, BV, N1W, N1B, W1P, B1P, W2P, B2P, N2W, N2B };

  // workspace bump allocator (all offsets 256B aligned)
  char* ws = (char*)d_ws;
  size_t off = 0;
  auto take = [&](size_t b) { size_t r = off; off = (off + b + 255) & ~(size_t)255; return r; };
  _Float16* ctx16   = (_Float16*)(ws + take((size_t)3 * L_ * CP * 2));
  _Float16* ctxcp16 = (_Float16*)(ws + take((size_t)3 * L_ * CP * 2));
  float*    qf      = (float*)   (ws + take((size_t)3 * L_ * CP * 4));
  _Float16* qh      = (_Float16*)(ws + take((size_t)L_ * CP * 2));
  _Float16* wk16    = (_Float16*)(ws + take((size_t)CP * CP * 2));
  _Float16* wv16    = (_Float16*)(ws + take((size_t)CP * CP * 2));
  _Float16* w116    = (_Float16*)(ws + take((size_t)CP * FP * 2));
  _Float16* w216    = (_Float16*)(ws + take((size_t)FP * CP * 2));
  float*    bkp     = (float*)   (ws + take((size_t)CP * 4));
  float*    bvp     = (float*)   (ws + take((size_t)CP * 4));
  float*    b1p     = (float*)   (ws + take((size_t)FP * 4));
  float*    b2p     = (float*)   (ws + take((size_t)CP * 4));
  float*    sA      = (float*)   (ws + take((size_t)L_ * CP * 4));
  float*    sB      = (float*)   (ws + take((size_t)L_ * CP * 4));
  _Float16* sH      = (_Float16*)(ws + take((size_t)L_ * FP * 2));
  float*    vcnt    = (float*)   (ws + take((size_t)3 * L_ * 4));
  float*    vsum    = sA;    // overlay: scatter phase predates GEMM scratch use
  float*    ctx_acc = qf;    // overlay: consumed before q is initialized

  auto Z = [&](float* p, long n) { zero_f32<<<dim3((unsigned)((n + 255) / 256)), 256, 0, stream>>>(p, n); };

  // --- phase 1: view contexts, mean over views ---
  Z(ctx_acc, 3L * L_ * C_);
  for (int v = 0; v < NV_; ++v) {
    Z(vsum, 3L * L_ * C_);
    Z(vcnt, 3L * L_);
    scatter_kernel<<<dim3((H_ * W_ + 255) / 256), 256, 0, stream>>>(
        feats + (size_t)v * C_ * H_ * W_, dep + (size_t)v * H_ * W_,
        c2w + (size_t)v * 16, intr + (size_t)v * 9, vsum, vcnt);
    ctx_accum_kernel<<<dim3((unsigned)((3L * L_ * C_ + 255) / 256)), 256, 0, stream>>>(vsum, vcnt, ctx_acc);
  }
  {
    long tot = 3L * L_ * CP;
    pad_f16_kernel<<<dim3((unsigned)((tot + 255) / 256)), 256, 0, stream>>>(ctx_acc, ctx16, 3 * L_, C_, CP, tot);
    pad_f32_kernel<<<dim3((unsigned)((tot + 255) / 256)), 256, 0, stream>>>(qpl, qf, 3 * L_, C_, CP, tot);
  }

  auto gemm = [&](const _Float16* Ap, const _Float16* Bp, const float* bp,
                  float* oF, _Float16* oH, int Kp, int Np, int gelu) {
    gemm_wmma<<<dim3(Np / 64, M_ / 256), 256, 0, stream>>>(Ap, Bp, bp, oF, oH, Kp, Np, gelu);
  };

  // --- phase 2: NL layers x (view-ctx block, cross-plane block) x 3 planes ---
  for (int i = 0; i < NL_; ++i) {
    for (int stage = 0; stage < 2; ++stage) {
      const float* const* P = stage ? cpp : cvp;
      {
        long t1 = (long)CP * CP;
        pad_f16_kernel<<<dim3((unsigned)((t1 + 255) / 256)), 256, 0, stream>>>(P[WK] + (size_t)i * C_ * C_, wk16, C_, C_, CP, t1);
        pad_f16_kernel<<<dim3((unsigned)((t1 + 255) / 256)), 256, 0, stream>>>(P[WV] + (size_t)i * C_ * C_, wv16, C_, C_, CP, t1);
        long t2 = (long)CP * FP;
        pad_f16_kernel<<<dim3((unsigned)((t2 + 255) / 256)), 256, 0, stream>>>(P[W1P] + (size_t)i * C_ * F_, w116, C_, F_, FP, t2);
        long t3 = (long)FP * CP;
        pad_f16_kernel<<<dim3((unsigned)((t3 + 255) / 256)), 256, 0, stream>>>(P[W2P] + (size_t)i * F_ * C_, w216, F_, C_, CP, t3);
        pad_bias_kernel<<<dim3((CP + 255) / 256), 256, 0, stream>>>(P[BK]  + (size_t)i * C_, bkp, C_, CP);
        pad_bias_kernel<<<dim3((CP + 255) / 256), 256, 0, stream>>>(P[BV]  + (size_t)i * C_, bvp, C_, CP);
        pad_bias_kernel<<<dim3((FP + 255) / 256), 256, 0, stream>>>(P[B1P] + (size_t)i * F_, b1p, F_, FP);
        pad_bias_kernel<<<dim3((CP + 255) / 256), 256, 0, stream>>>(P[B2P] + (size_t)i * C_, b2p, C_, CP);
      }
      if (stage == 1)  // cross-plane context from all updated planes
        ctxcp_kernel<<<dim3(L_, 3), 256, 0, stream>>>(qf, ctxcp16);
      for (int p = 0; p < 3; ++p) {
        const _Float16* Actx = (stage ? ctxcp16 : ctx16) + (size_t)p * L_ * CP;
        float* qfp = qf + (size_t)p * L_ * CP;
        gemm(Actx, wk16, bkp, sA, nullptr, CP, CP, 0);                 // k = ctx@Wk + bk
        gemm(Actx, wv16, bvp, sB, nullptr, CP, CP, 0);                 // v = ctx@Wv + bv
        ln_mid_kernel<<<dim3(L_), 256, 0, stream>>>(qfp, qh, sA, sB,
            P[LNW] + (size_t)i * C_, P[LNB] + (size_t)i * C_,
            P[N1W] + (size_t)i * C_, P[N1B] + (size_t)i * C_);
        gemm(qh, w116, b1p, nullptr, sH, CP, FP, 1);                   // h1 = gelu(q@W1+b1)
        gemm(sH, w216, b2p, sA, nullptr, FP, CP, 0);                   // h2 = h1@W2 + b2
        ln_post_kernel<<<dim3(L_), 256, 0, stream>>>(qfp, sA,
            P[N2W] + (size_t)i * C_, P[N2B] + (size_t)i * C_);
      }
    }
  }
  copy_out_kernel<<<dim3((unsigned)((3L * L_ * C_ + 255) / 256)), 256, 0, stream>>>(qf, (float*)d_out);
}
